// CausalSelfAttention_27333171872552
// MI455X (gfx1250) — matrix operand order repaired and checked by reading, not compiled, not run on hardware
//
#include <hip/hip_runtime.h>
#include <hip/hip_bf16.h>

// ---------------------------------------------------------------------------
// CausalSelfAttention forward for MI455X (gfx1250), wave32 + WMMA bf16.
// B=2, T=2048, C=1024, H=16, HD=64.
// Pipeline: f32->bf16 convert -> QKV WMMA GEMM (async-to-LDS staged B,
// double buffered, 16x64 wave tile, K-loop not unrolled to keep accumulators
// pinned) -> flash attention (WMMA, online softmax, NT h-stream)
// -> output projection WMMA GEMM (f32 out).
// ---------------------------------------------------------------------------

#define Bq 2
#define Tq 2048
#define Cq 1024
#define Hq 16
#define HDq 64
#define BT (Bq * Tq)

#define GLOBAL_AS __attribute__((address_space(1)))
#define LDS_AS    __attribute__((address_space(3)))

typedef __attribute__((ext_vector_type(16))) __bf16 v16bf;
typedef __attribute__((ext_vector_type(8)))  __bf16 v8bf;
typedef __attribute__((ext_vector_type(8)))  float  v8f;
typedef __attribute__((ext_vector_type(4)))  int    v4i;

#define WMMA_BF16(a, b, c) \
  __builtin_amdgcn_wmma_f32_16x16x32_bf16(false, (a), false, (b), (short)0, (c), false, false)

#if __has_builtin(__builtin_amdgcn_global_load_async_to_lds_b128)
#define HAVE_ASYNC_LDS 1
#else
#define HAVE_ASYNC_LDS 0
#endif

__device__ __forceinline__ void wait_async_zero() {
#if HAVE_ASYNC_LDS
#if __has_builtin(__builtin_amdgcn_s_wait_asynccnt)
  __builtin_amdgcn_s_wait_asynccnt(0);
#else
  asm volatile("s_wait_asynccnt 0x0" ::: "memory");
#endif
#endif
}

// A-fragment (16x32, bf16) from row-major [rows x ldk]:
//   lane 0-15 : row = lane,    elems 0..15 = K 0..15
//   lane 16-31: row = lane-16, elems 0..15 = K 16..31
__device__ __forceinline__ v16bf load_a_frag(const __bf16* A, int ldk,
                                             int row0, int k0, int lane) {
  int row = row0 + (lane & 15);
  int kb  = k0 + ((lane >> 4) << 4);
  const __bf16* p = A + (size_t)row * ldk + kb;
  v8bf lo = *(const v8bf*)(p);
  v8bf hi = *(const v8bf*)(p + 8);
  v16bf r;
#pragma unroll
  for (int i = 0; i < 8; ++i) { r[i] = lo[i]; r[i + 8] = hi[i]; }
  return r;
}

// B-fragment (32x16, bf16) where the source matrix is stored [N x K] row-major
// (B[k][n] = src[n][k]): lane 0-15: n=lane, K 0..15; lane 16-31: n=lane-16, K 16..31
__device__ __forceinline__ v16bf load_b_frag_nk(const __bf16* S, int ldk,
                                                int n0, int k0, int lane) {
  int n  = n0 + (lane & 15);
  int kb = k0 + ((lane >> 4) << 4);
  const __bf16* p = S + (size_t)n * ldk + kb;
  v8bf lo = *(const v8bf*)(p);
  v8bf hi = *(const v8bf*)(p + 8);
  v16bf r;
#pragma unroll
  for (int i = 0; i < 8; ++i) { r[i] = lo[i]; r[i + 8] = hi[i]; }
  return r;
}

// Same B-fragment but from an LDS tile [64 rows x 40 stride], K-tile of 32.
__device__ __forceinline__ v16bf load_b_frag_lds(const __bf16* base, int n0, int lane) {
  int n  = n0 + (lane & 15);
  int kb = (lane >> 4) << 4;
  const __bf16* p = base + n * 40 + kb;
  v8bf lo = *(const v8bf*)(p);
  v8bf hi = *(const v8bf*)(p + 8);
  v16bf r;
#pragma unroll
  for (int i = 0; i < 8; ++i) { r[i] = lo[i]; r[i + 8] = hi[i]; }
  return r;
}

// Stage one 64(N) x 32(K) bf16 weight tile into LDS (row stride 40 elems).
// 256 threads, 16 B each: thread -> (row = tid>>2, chunk = tid&3).
__device__ __forceinline__ void stage_B(const __bf16* W, int nbase, int kb,
                                        __bf16* dst) {
  int tid   = threadIdx.x;
  int row   = tid >> 2;
  int chunk = tid & 3;
  const __bf16* g = W + (size_t)(nbase + row) * Cq + kb + chunk * 8;
  __bf16* l = dst + row * 40 + chunk * 8;
#if HAVE_ASYNC_LDS
  __builtin_amdgcn_global_load_async_to_lds_b128((GLOBAL_AS v4i*)g,
                                                 (LDS_AS v4i*)l, 0, 0);
#else
  *(v8bf*)l = *(const v8bf*)g;
#endif
}

// 16(M) x 64(N) per-wave GEMM tile, K = Cq, B staged in double-buffered LDS
// (shared by all 8 waves of the block). K-loop deliberately NOT unrolled so
// the 4 accumulators keep a single physical register assignment (no v_mov
// rotation, no WMMA->VALU hazard nops).
__device__ __forceinline__ void gemm_16x64(const __bf16* __restrict__ A,
                                           const __bf16* __restrict__ W,
                                           int mbase, int nbase,
                                           __bf16* sB,      // [2][64*40]
                                           v8f acc[4]) {
  int lane = threadIdx.x & 31;
  stage_B(W, nbase, 0, sB);
#pragma clang loop unroll(disable)
  for (int i = 0; i < Cq / 32; ++i) {
    int kb = i * 32;
    wait_async_zero();
    __syncthreads();
    if (i + 1 < Cq / 32) stage_B(W, nbase, kb + 32, sB + ((i + 1) & 1) * 64 * 40);
    const __bf16* buf = sB + (i & 1) * 64 * 40;
    v16bf a = load_a_frag(A, Cq, mbase, kb, lane);
#pragma unroll
    for (int nt = 0; nt < 4; ++nt) {
      v16bf b = load_b_frag_lds(buf, nt * 16, lane);
      acc[nt] = WMMA_BF16(a, b, acc[nt]);
    }
  }
}

// ---------------------------------------------------------------------------
__global__ void cvt_f32_bf16(const float* __restrict__ in,
                             __bf16* __restrict__ out, int n) {
  int i = blockIdx.x * blockDim.x + threadIdx.x;
  if (i < n) out[i] = (__bf16)in[i];
}

// ---------------------------------------------------------------------------
// y = x @ W^T + bias -> [B,H,T,HD] bf16 (head split), q pre-scaled by 1/8.
__global__ void qkv_gemm(const __bf16* __restrict__ xb,
                         const __bf16* __restrict__ w,
                         const float* __restrict__ bias,
                         __bf16* __restrict__ out, float scale) {
  __shared__ __bf16 sB[2][64 * 40];
  int lane  = threadIdx.x & 31;
  int wid   = threadIdx.x >> 5;
  int mbase = blockIdx.x * 128 + wid * 16;
  int nbase = blockIdx.y * 64;

  v8f acc[4] = {};
  gemm_16x64(xb, w, mbase, nbase, &sB[0][0], acc);

  int colh  = lane & 15;
  int rbase = (lane >> 4) * 8;
#pragma unroll
  for (int nt = 0; nt < 4; ++nt) {
    int n = nbase + nt * 16 + colh;
    float bv = bias[n];
    int hh = n >> 6, d = n & 63;
#pragma unroll
    for (int r = 0; r < 8; ++r) {
      int m = mbase + rbase + r;
      int b = m >> 11, t = m & (Tq - 1);
      float val = (acc[nt][r] + bv) * scale;
      out[(((size_t)b * Hq + hh) * Tq + t) * HDq + d] = (__bf16)val;
    }
  }
}

// ---------------------------------------------------------------------------
// Flash attention: one wave owns 16 query rows of one (b,h). Key blocks of 32.
// q is pre-scaled by 1/8; h streamed with non-temporal loads (read-once).
__global__ void flash_attn(const __bf16* __restrict__ q,
                           const __bf16* __restrict__ k,
                           const __bf16* __restrict__ v,
                           const float* __restrict__ hbias,
                           __bf16* __restrict__ yatt) {
  int lane  = threadIdx.x & 31;
  int wid   = threadIdx.x >> 5;
  int bh    = blockIdx.y;                      // b*H + h
  int qbase = blockIdx.x * 128 + wid * 16;

  const __bf16* qh = q + (size_t)bh * Tq * HDq;
  const __bf16* kh = k + (size_t)bh * Tq * HDq;
  const __bf16* vh = v + (size_t)bh * Tq * HDq;
  const float*  hp = hbias + (size_t)bh * Tq * Tq;

  // per-wave P staging tile: 16 rows x 32 cols, stride 40 (pad vs banks)
  __shared__ __bf16 sP[8][16 * 40];
  __bf16* myP = &sP[wid][0];

  v16bf aQ[2];
#pragma unroll
  for (int c = 0; c < 2; ++c) aQ[c] = load_a_frag(qh, HDq, qbase, c * 32, lane);

  float mrow[8], lrow[8];
#pragma unroll
  for (int r = 0; r < 8; ++r) { mrow[r] = -1e30f; lrow[r] = 0.0f; }
  v8f accO[4] = {};

  int rbase = (lane >> 4) * 8;
  int colh  = lane & 15;

#pragma clang loop unroll(disable)
  for (int j0 = 0; j0 < qbase + 16; j0 += 32) {
    // ---- S = (q/8) . K^T  (16 queries x 32 keys) ----
    v8f S[2];
#pragma unroll
    for (int nt = 0; nt < 2; ++nt) {
      v8f s = {};
#pragma unroll
      for (int c = 0; c < 2; ++c) {
        v16bf bk = load_b_frag_nk(kh, HDq, j0 + nt * 16, c * 32, lane);
        s = WMMA_BF16(aQ[c], bk, s);
      }
      S[nt] = s;
    }

    // ---- + h, causal mask, online softmax (rows live in 16-lane halves) ----
    float p0a[8], p1a[8];
#pragma unroll
    for (int r = 0; r < 8; ++r) {
      int m = qbase + rbase + r;
      const float* hr = hp + (size_t)m * Tq;
      int n0 = j0 + colh, n1 = j0 + 16 + colh;
      float s0 = S[0][r] + __builtin_nontemporal_load(hr + n0);
      float s1 = S[1][r] + __builtin_nontemporal_load(hr + n1);
      if (n0 > m) s0 = -1e30f;
      if (n1 > m) s1 = -1e30f;
      float mx = fmaxf(s0, s1);
#pragma unroll
      for (int off = 1; off < 16; off <<= 1)
        mx = fmaxf(mx, __shfl_xor(mx, off, 32));
      float mnew  = fmaxf(mrow[r], mx);
      float alpha = __expf(mrow[r] - mnew);
      float p0 = __expf(s0 - mnew);
      float p1 = __expf(s1 - mnew);
      float ps = p0 + p1;
#pragma unroll
      for (int off = 1; off < 16; off <<= 1) ps += __shfl_xor(ps, off, 32);
      lrow[r] = lrow[r] * alpha + ps;
      mrow[r] = mnew;
#pragma unroll
      for (int t = 0; t < 4; ++t) accO[t][r] *= alpha;
      p0a[r] = p0; p1a[r] = p1;
    }

    // ---- P: C-layout -> LDS (row-major 16x32) -> A-fragment ----
#pragma unroll
    for (int r = 0; r < 8; ++r) {
      myP[(rbase + r) * 40 + colh]      = (__bf16)p0a[r];
      myP[(rbase + r) * 40 + 16 + colh] = (__bf16)p1a[r];
    }
    v16bf aP;
    {
      int row = lane & 15;
      int kb  = (lane >> 4) << 4;
      const __bf16* pp = myP + row * 40 + kb;
      v8bf lo = *(const v8bf*)pp;
      v8bf hi = *(const v8bf*)(pp + 8);
#pragma unroll
      for (int i = 0; i < 8; ++i) { aP[i] = lo[i]; aP[i + 8] = hi[i]; }
    }

    // ---- O += P @ V : B[kk][n] = v[j0+kk][dt*16+n] (per-lane gather) ----
#pragma unroll
    for (int t = 0; t < 4; ++t) {
      int d   = t * 16 + colh;
      int kkb = (lane >> 4) << 4;
      const __bf16* vp = vh + (size_t)(j0 + kkb) * HDq + d;
      v16bf bv;
#pragma unroll
      for (int e = 0; e < 16; ++e) bv[e] = vp[(size_t)e * HDq];
      accO[t] = WMMA_BF16(aP, bv, accO[t]);
    }
  }

  // ---- normalize and store to [B*T, C] bf16 (heads re-merged) ----
  int b  = bh >> 4;
  int hh = bh & 15;
#pragma unroll
  for (int t = 0; t < 4; ++t) {
    int col = hh * HDq + t * 16 + colh;
#pragma unroll
    for (int r = 0; r < 8; ++r) {
      int trow  = qbase + rbase + r;
      float val = accO[t][r] / lrow[r];
      yatt[((size_t)b * Tq + trow) * Cq + col] = (__bf16)val;
    }
  }
}

// ---------------------------------------------------------------------------
// out = yatt @ Wp^T + bp  -> fp32 [B*T, C]
__global__ void proj_out(const __bf16* __restrict__ yb,
                         const __bf16* __restrict__ wp,
                         const float* __restrict__ bias,
                         float* __restrict__ out) {
  __shared__ __bf16 sB[2][64 * 40];
  int lane  = threadIdx.x & 31;
  int wid   = threadIdx.x >> 5;
  int mbase = blockIdx.x * 128 + wid * 16;
  int nbase = blockIdx.y * 64;

  v8f acc[4] = {};
  gemm_16x64(yb, wp, mbase, nbase, &sB[0][0], acc);

  int colh  = lane & 15;
  int rbase = (lane >> 4) * 8;
#pragma unroll
  for (int nt = 0; nt < 4; ++nt) {
    int n = nbase + nt * 16 + colh;
    float bv = bias[n];
#pragma unroll
    for (int r = 0; r < 8; ++r) {
      int m = mbase + rbase + r;
      out[(size_t)m * Cq + n] = acc[nt][r] + bv;
    }
  }
}

// ---------------------------------------------------------------------------
extern "C" void kernel_launch(void* const* d_in, const int* in_sizes, int n_in,
                              void* d_out, int out_size, void* d_ws, size_t ws_size,
                              hipStream_t stream) {
  const float* x  = (const float*)d_in[0];
  const float* h  = (const float*)d_in[1];
  const float* Wq = (const float*)d_in[2];
  const float* bq = (const float*)d_in[3];
  const float* Wk = (const float*)d_in[4];
  const float* bk = (const float*)d_in[5];
  const float* Wv = (const float*)d_in[6];
  const float* bv = (const float*)d_in[7];
  const float* Wp = (const float*)d_in[8];
  const float* bp = (const float*)d_in[9];
  float* out = (float*)d_out;

  const size_t MB = 1024ull * 1024ull;
  char* ws = (char*)d_ws;
  __bf16* xb  = (__bf16*)(ws + 0 * MB);   // 8 MB  [B*T, C]
  __bf16* wqb = (__bf16*)(ws + 8 * MB);   // 2 MB
  __bf16* wkb = (__bf16*)(ws + 10 * MB);  // 2 MB
  __bf16* wvb = (__bf16*)(ws + 12 * MB);  // 2 MB
  __bf16* wpb = (__bf16*)(ws + 14 * MB);  // 2 MB
  __bf16* qb  = (__bf16*)(ws + 16 * MB);  // 8 MB  [B,H,T,HD], pre-scaled 1/8
  __bf16* kb  = (__bf16*)(ws + 24 * MB);  // 8 MB
  __bf16* vb  = (__bf16*)(ws + 32 * MB);  // 8 MB
  __bf16* yb  = (__bf16*)(ws + 40 * MB);  // 8 MB  [B*T, C]  (total 48 MB)

  const int nX = BT * Cq;       // 4,194,304
  const int nW = Cq * Cq;       // 1,048,576
  cvt_f32_bf16<<<(nX + 255) / 256, 256, 0, stream>>>(x,  xb,  nX);
  cvt_f32_bf16<<<(nW + 255) / 256, 256, 0, stream>>>(Wq, wqb, nW);
  cvt_f32_bf16<<<(nW + 255) / 256, 256, 0, stream>>>(Wk, wkb, nW);
  cvt_f32_bf16<<<(nW + 255) / 256, 256, 0, stream>>>(Wv, wvb, nW);
  cvt_f32_bf16<<<(nW + 255) / 256, 256, 0, stream>>>(Wp, wpb, nW);

  dim3 gemmGrid(BT / 128, Cq / 64);           // 32 x 16
  qkv_gemm<<<gemmGrid, 256, 0, stream>>>(xb, wqb, bq, qb, 0.125f);
  qkv_gemm<<<gemmGrid, 256, 0, stream>>>(xb, wkb, bk, kb, 1.0f);
  qkv_gemm<<<gemmGrid, 256, 0, stream>>>(xb, wvb, bv, vb, 1.0f);

  dim3 attnGrid(Tq / 128, Bq * Hq);           // 16 x 32
  flash_attn<<<attnGrid, 256, 0, stream>>>(qb, kb, vb, h, yb);

  proj_out<<<gemmGrid, 256, 0, stream>>>(yb, wpb, bp, out);
}
